// GlobalPointer_53463752901334
// MI455X (gfx1250) — compile-verified, matrix operand order checked by reading, not run on hardware
//
#include <hip/hip_runtime.h>
#include <hip/hip_bf16.h>

typedef __attribute__((ext_vector_type(16))) _Float16 v16h;
typedef __attribute__((ext_vector_type(8)))  _Float16 v8h;
typedef __attribute__((ext_vector_type(4)))  _Float16 v4h;
typedef __attribute__((ext_vector_type(8)))  float    v8f;
typedef __attribute__((ext_vector_type(4)))  unsigned int u32x4;
typedef __attribute__((ext_vector_type(8)))  int      i32x8;
typedef __attribute__((ext_vector_type(4)))  int      i32x4;

#define HIDDEN 1024
#define OUTD   1536
#define MROWS  16384   // B*S
#define NHEADS 12
#define HEADSZ 64
#define SEQ    512
#define BATCHN 32

// Async DMA: global -> LDS, 16B per lane, tracked by ASYNCcnt (no VGPR data path).
// GVS addressing: mem_addr = SADDR(64) + VADDR(32, bytes) + IOFFSET.
__device__ __forceinline__ void async_g2l_b128(unsigned lds_addr, unsigned goff,
                                               const void* sbase) {
  asm volatile("global_load_async_to_lds_b128 %0, %1, %2 offset:0"
               :: "v"(lds_addr), "v"(goff), "s"(sbase) : "memory");
}
__device__ __forceinline__ void wait_async0() {
  asm volatile("s_wait_asynccnt 0" ::: "memory");
}

// Tensor Data Mover: DMA a 2D f16 tile (tileW x tileH, row stride strideElem,
// tensor bounds dim0 x dim1 elements) from global into LDS. One call per wave.
// 6-arg builtin form (amdgpu-toolchain / clang-23): (g0, g1, g2, g3, g?, cpol).
__device__ __forceinline__ void tdm_load_tile_f16(unsigned lds_addr,
                                                  const void* gtile,
                                                  unsigned tileW, unsigned tileH,
                                                  unsigned dim0, unsigned dim1,
                                                  unsigned strideElem) {
  unsigned long long ga = (unsigned long long)(size_t)gtile;
  u32x4 g0;
  g0[0] = 1u;                                   // count=1, user mode, no gather
  g0[1] = lds_addr;                             // D#.lds_addr (bytes)
  g0[2] = (unsigned)ga;                         // global_addr[31:0]
  g0[3] = 0x80000000u | (unsigned)(ga >> 32);   // type=2 | global_addr[56:32]
  i32x8 g1;
  g1[0] = 0x00010000;                           // data_size=1 (2 bytes/elem)
  g1[1] = (int)((dim0 & 0xFFFFu) << 16);        // tensor_dim0[15:0]
  g1[2] = (int)((dim0 >> 16) | ((dim1 & 0xFFFFu) << 16));   // dim0 hi | dim1 lo
  g1[3] = (int)((dim1 >> 16) | (tileW << 16));  // dim1 hi | tile_dim0
  g1[4] = (int)tileH;                           // tile_dim1 (tile_dim2=0)
  g1[5] = (int)strideElem;                      // tensor_dim0_stride[31:0]
  g1[6] = 0;                                    // stride hi | dim1_stride lo
  g1[7] = 0;
  i32x4 z4 = {0, 0, 0, 0};                      // 2D tile: groups 2/3 unused
  i32x8 z8 = {0, 0, 0, 0, 0, 0, 0, 0};
  __builtin_amdgcn_tensor_load_to_lds(g0, g1, z4, z4, z8, 0);
}

// ---------------------------------------------------------------------------
// Kernel 1a: convert x (f32) -> xh (f16), 4 elements/thread
// ---------------------------------------------------------------------------
__global__ __launch_bounds__(256) void cvt_x_f16(const float* __restrict__ x,
                                                 _Float16* __restrict__ xh, int n4) {
  int i = blockIdx.x * blockDim.x + threadIdx.x;
  if (i >= n4) return;
  float4 f = ((const float4*)x)[i];
  v4h h;
  h[0] = (_Float16)f.x; h[1] = (_Float16)f.y;
  h[2] = (_Float16)f.z; h[3] = (_Float16)f.w;
  *(v4h*)(xh + 4 * (size_t)i) = h;
}

// ---------------------------------------------------------------------------
// Kernel 1b: W [K=1024][N=1536] f32 -> wt [N][K] f16 (transposed)
// ---------------------------------------------------------------------------
__global__ __launch_bounds__(256) void cvt_w_t(const float* __restrict__ W,
                                               _Float16* __restrict__ wt, int total) {
  int i = blockIdx.x * blockDim.x + threadIdx.x; // linear over N*K, k fastest
  if (i >= total) return;
  int n = i >> 10;        // /1024
  int k = i & 1023;
  wt[i] = (_Float16)W[(size_t)k * OUTD + n];
}

// ---------------------------------------------------------------------------
// Kernel 2: h = xh * W, WMMA f16->f32, 128x128 tile, BK=32, double-buffered.
// A-panel via per-lane async DMA (ASYNCcnt); B-panel via one TDM descriptor
// per k-step (TENSORcnt). 8 waves: 4 (M) x 2 (N), 2x4 tiles each.
// ---------------------------------------------------------------------------
__global__ __launch_bounds__(256) void wmma_gemm_h(const _Float16* __restrict__ xh,
                                                   const _Float16* __restrict__ wt,
                                                   float* __restrict__ hbuf) {
  __shared__ _Float16 As[2][128][32];   // 8 KB per buffer
  __shared__ _Float16 Bs[2][128][32];   // (wt rows are matrix columns)

  const int tid  = threadIdx.x;
  const int lane = tid & 31;
  const int wave = tid >> 5;
  const int wr   = wave & 3;            // M quadrant (32 rows each)
  const int wc   = wave >> 2;           // N half    (64 cols each)
  const int mBase = blockIdx.y * 128;
  const int nBase = blockIdx.x * 128;
  const int lrow  = lane & 15;
  const int kb    = (lane >> 4) * 8;    // ISA 16-bit A/B layout: lanes 16-31 -> K+8

  v8f acc[2][4] = {};

  // A staging: 512 16B chunks, 2 per thread, async global->LDS.
  auto stageA = [&](int buf, int kt) {
    const int k0 = kt * 32;
#pragma unroll
    for (int p = 0; p < 2; ++p) {
      int c   = tid + p * 256;          // 0..511
      int row = c >> 2;                 // 0..127
      int seg = (c & 3) * 8;            // half offset 0/8/16/24
      unsigned ldsA = (unsigned)(size_t)&As[buf][row][seg];
      unsigned ga = (unsigned)(((unsigned)(mBase + row) * HIDDEN + k0 + seg) * 2u);
      async_g2l_b128(ldsA, ga, xh);
    }
  };
  // B staging: one 128x32 f16 tile from wt, single TDM op issued by wave 0.
  auto stageB = [&](int buf, int kt) {
    if (wave == 0) {
      unsigned lds = (unsigned)(size_t)&Bs[buf][0][0];
      const _Float16* gtile = wt + (size_t)nBase * HIDDEN + kt * 32;
      tdm_load_tile_f16(lds, gtile, /*tileW=*/32, /*tileH=*/128,
                        /*dim0=*/HIDDEN, /*dim1=*/OUTD, /*stride=*/HIDDEN);
    }
  };

  stageA(0, 0);
  stageB(0, 0);
  wait_async0();
  __builtin_amdgcn_s_wait_tensorcnt(0);
  __syncthreads();

  for (int kt = 0; kt < 32; ++kt) {
    const int cur = kt & 1;
    if (kt + 1 < 32) {
      stageA(cur ^ 1, kt + 1);          // DMA overlaps the WMMA below
      stageB(cur ^ 1, kt + 1);
      __builtin_prefetch(xh + (size_t)(mBase + (tid >> 1)) * HIDDEN + (kt + 2) * 32, 0, 1);
    }

    v16h a[2], b[4];
#pragma unroll
    for (int sm = 0; sm < 2; ++sm) {
      const int row = wr * 32 + sm * 16 + lrow;
      v8h lo = *(const v8h*)(&As[cur][row][kb]);
      v8h hi = *(const v8h*)(&As[cur][row][kb + 16]);
      a[sm] = __builtin_shufflevector(lo, hi, 0,1,2,3,4,5,6,7,8,9,10,11,12,13,14,15);
    }
#pragma unroll
    for (int sn = 0; sn < 4; ++sn) {
      const int col = wc * 64 + sn * 16 + lrow;
      v8h lo = *(const v8h*)(&Bs[cur][col][kb]);
      v8h hi = *(const v8h*)(&Bs[cur][col][kb + 16]);
      b[sn] = __builtin_shufflevector(lo, hi, 0,1,2,3,4,5,6,7,8,9,10,11,12,13,14,15);
    }

#pragma unroll
    for (int sm = 0; sm < 2; ++sm)
#pragma unroll
      for (int sn = 0; sn < 4; ++sn)
        acc[sm][sn] = __builtin_amdgcn_wmma_f32_16x16x32_f16(
            false, a[sm], false, b[sn], (short)0, acc[sm][sn], false, false);

    wait_async0();                      // A-panel DMA complete
    __builtin_amdgcn_s_wait_tensorcnt(0); // B-panel TDM complete (no-op off wave 0)
    __syncthreads();
  }

  // C/D layout: VGPR r, lanes 0-15 -> M=r, lanes 16-31 -> M=8+r; N = lane%16
  const int rowOff = (lane >> 4) * 8;
#pragma unroll
  for (int sm = 0; sm < 2; ++sm)
    for (int sn = 0; sn < 4; ++sn)
#pragma unroll
      for (int r = 0; r < 8; ++r) {
        int row = mBase + wr * 32 + sm * 16 + rowOff + r;
        int col = nBase + wc * 64 + sn * 16 + lrow;
        hbuf[(size_t)row * OUTD + col] = acc[sm][sn][r];
      }
}

// ---------------------------------------------------------------------------
// Kernel 3: bias + LayerNorm(1536) + diagonal "RoPE" scale + head split.
// One block per row; q gets the 1/sqrt(64)=0.125 fold. Writes q,v f16 as
// [B, H, S, 64] so the score-GEMM fragment loads are contiguous.
// ---------------------------------------------------------------------------
__global__ __launch_bounds__(256) void ln_rope_split(const float* __restrict__ hbuf,
                                                     const float* __restrict__ bias,
                                                     const float* __restrict__ gamma,
                                                     const float* __restrict__ beta,
                                                     _Float16* __restrict__ qb,
                                                     _Float16* __restrict__ vb) {
  const int row  = blockIdx.x;          // 0..16383
  const int tid  = threadIdx.x;
  const int lane = tid & 31;
  const int wave = tid >> 5;
  __shared__ float red[8];

  float vals[6];
  float s = 0.f;
#pragma unroll
  for (int e = 0; e < 6; ++e) {
    int c = tid + e * 256;
    float t = hbuf[(size_t)row * OUTD + c] + bias[c];
    vals[e] = t;
    s += t;
  }
  for (int o = 16; o > 0; o >>= 1) s += __shfl_xor(s, o, 32);
  if (lane == 0) red[wave] = s;
  __syncthreads();
  float tot = 0.f;
#pragma unroll
  for (int w = 0; w < 8; ++w) tot += red[w];
  const float mean = tot * (1.f / OUTD);
  __syncthreads();

  float s2 = 0.f;
#pragma unroll
  for (int e = 0; e < 6; ++e) { float d = vals[e] - mean; s2 += d * d; }
  for (int o = 16; o > 0; o >>= 1) s2 += __shfl_xor(s2, o, 32);
  if (lane == 0) red[wave] = s2;
  __syncthreads();
  float tot2 = 0.f;
#pragma unroll
  for (int w = 0; w < 8; ++w) tot2 += red[w];
  const float rstd = rsqrtf(tot2 * (1.f / OUTD) + 1e-5f);

  const int b = row >> 9;               // row / 512
  const int sq = row & 511;
#pragma unroll
  for (int e = 0; e < 6; ++e) {
    int c = tid + e * 256;
    float g = (vals[e] - mean) * rstd * gamma[c] + beta[c];
    int head = c >> 7;
    int r = c & 127;
    int j = (r < 64) ? r : (r - 64);
    // theta_j = 10000^(-2*(j/2)/64); scale = cos(theta) + (j even ? -1 : +1)*sin(theta)
    float theta = __expf(-0.28782082f * (float)(j >> 1)); // ln(1e4)/32
    float sc = __cosf(theta) + ((j & 1) ? __sinf(theta) : -__sinf(theta));
    size_t dst = ((size_t)(b * NHEADS + head) * SEQ + sq) * HEADSZ + j;
    if (r < 64) qb[dst] = (_Float16)(g * sc * 0.125f);
    else        vb[dst] = (_Float16)(g * sc);
  }
}

// ---------------------------------------------------------------------------
// Kernel 4: scores[b,h,i,j] = sum_d q[i,d]*v[j,d]   (WMMA, K=64 = 2 k-steps)
// 128x128 output tile per block; staging via async global->LDS DMA.
// ---------------------------------------------------------------------------
__global__ __launch_bounds__(256) void attn_scores_wmma(const _Float16* __restrict__ qb,
                                                        const _Float16* __restrict__ vb,
                                                        float* __restrict__ out) {
  __shared__ _Float16 qs[128][64];      // 16 KB
  __shared__ _Float16 vs[128][64];      // 16 KB

  const int tid  = threadIdx.x;
  const int lane = tid & 31;
  const int wave = tid >> 5;
  const int wr   = wave & 3;
  const int wc   = wave >> 2;
  const int lrow = lane & 15;
  const int kb   = (lane >> 4) * 8;
  const int batch = blockIdx.z;         // b*12+h
  const int i0 = blockIdx.y * 128;
  const int j0 = blockIdx.x * 128;
  const unsigned base = (unsigned)batch * SEQ * HEADSZ * 2u;  // byte offset

#pragma unroll
  for (int p = 0; p < 4; ++p) {
    int c   = tid + p * 256;            // 0..1023 chunks of 16B
    int row = c >> 3;                   // 0..127
    int seg = (c & 7) * 8;
    unsigned ldsQ = (unsigned)(size_t)&qs[row][seg];
    unsigned ldsV = (unsigned)(size_t)&vs[row][seg];
    unsigned gq = base + ((unsigned)(i0 + row) * HEADSZ + seg) * 2u;
    unsigned gv = base + ((unsigned)(j0 + row) * HEADSZ + seg) * 2u;
    async_g2l_b128(ldsQ, gq, qb);
    async_g2l_b128(ldsV, gv, vb);
  }
  wait_async0();
  __syncthreads();

  v8f acc[2][4] = {};
#pragma unroll
  for (int ks = 0; ks < 2; ++ks) {
    const int k0 = ks * 32;
    v16h a[2], b[4];
#pragma unroll
    for (int sm = 0; sm < 2; ++sm) {
      const int row = wr * 32 + sm * 16 + lrow;
      v8h lo = *(const v8h*)(&qs[row][k0 + kb]);
      v8h hi = *(const v8h*)(&qs[row][k0 + kb + 16]);
      a[sm] = __builtin_shufflevector(lo, hi, 0,1,2,3,4,5,6,7,8,9,10,11,12,13,14,15);
    }
#pragma unroll
    for (int sn = 0; sn < 4; ++sn) {
      const int col = wc * 64 + sn * 16 + lrow;
      v8h lo = *(const v8h*)(&vs[col][k0 + kb]);
      v8h hi = *(const v8h*)(&vs[col][k0 + kb + 16]);
      b[sn] = __builtin_shufflevector(lo, hi, 0,1,2,3,4,5,6,7,8,9,10,11,12,13,14,15);
    }
#pragma unroll
    for (int sm = 0; sm < 2; ++sm)
#pragma unroll
      for (int sn = 0; sn < 4; ++sn)
        acc[sm][sn] = __builtin_amdgcn_wmma_f32_16x16x32_f16(
            false, a[sm], false, b[sn], (short)0, acc[sm][sn], false, false);
  }

  const int rowOff = (lane >> 4) * 8;
  const size_t ob = (size_t)batch * SEQ * SEQ;
#pragma unroll
  for (int sm = 0; sm < 2; ++sm)
    for (int sn = 0; sn < 4; ++sn)
#pragma unroll
      for (int r = 0; r < 8; ++r) {
        int row = i0 + wr * 32 + sm * 16 + rowOff + r;
        int col = j0 + wc * 64 + sn * 16 + lrow;
        out[ob + (size_t)row * SEQ + col] = acc[sm][sn][r];
      }
}

// ---------------------------------------------------------------------------
extern "C" void kernel_launch(void* const* d_in, const int* in_sizes, int n_in,
                              void* d_out, int out_size, void* d_ws, size_t ws_size,
                              hipStream_t stream) {
  const float* x     = (const float*)d_in[0];
  const float* W     = (const float*)d_in[1];
  const float* bias  = (const float*)d_in[2];
  const float* gamma = (const float*)d_in[3];
  const float* beta  = (const float*)d_in[4];
  float* out = (float*)d_out;

  // workspace layout (all offsets 256B-aligned)
  char* w = (char*)d_ws;
  _Float16* xh = (_Float16*)w;                                     // 33,554,432 B
  _Float16* wt = (_Float16*)(w + (size_t)33554432);                //  3,145,728 B
  float*    hb = (float*)   (w + (size_t)36700160);                // 100,663,296 B
  _Float16* qb = (_Float16*)(w + (size_t)137363456);               // 25,165,824 B
  _Float16* vb = (_Float16*)(w + (size_t)162529280);               // 25,165,824 B

  // 1) precision staging
  cvt_x_f16<<<(MROWS * HIDDEN / 4 + 255) / 256, 256, 0, stream>>>(x, xh, MROWS * HIDDEN / 4);
  cvt_w_t<<<(OUTD * HIDDEN + 255) / 256, 256, 0, stream>>>(W, wt, OUTD * HIDDEN);

  // 2) h = x*W  (bias folded into LN kernel)
  dim3 g1(OUTD / 128, MROWS / 128);   // (12, 128)
  wmma_gemm_h<<<g1, 256, 0, stream>>>(xh, wt, hb);

  // 3) bias + LN + rope-scale + split/transpose
  ln_rope_split<<<MROWS, 256, 0, stream>>>(hb, bias, gamma, beta, qb, vb);

  // 4) scores = q @ v^T (scaled)
  dim3 g2(SEQ / 128, SEQ / 128, BATCHN * NHEADS); // (4, 4, 384)
  attn_scores_wmma<<<g2, 256, 0, stream>>>(qb, vb, out);
}
